// QuantumFeedForward_65481071408116
// MI455X (gfx1250) — compile-verified, matrix operand order checked by reading, not run on hardware
//
#include <hip/hip_runtime.h>
#include <math.h>

typedef __attribute__((ext_vector_type(2)))  float     v2f;
typedef __attribute__((ext_vector_type(4)))  float     v4f;
typedef __attribute__((ext_vector_type(8)))  float     v8f;
typedef __attribute__((ext_vector_type(16))) _Float16  v16h;

#define B_DIM  8192
#define E_DIM  2048
#define FF_DIM 8192

// ---------------------------------------------------------------------------
// Phase 1: row[e] = relu(cos(q0+q1)) * (W2[e,0]+W2[e,1]+W2[e,2]+W2[e,3]) + b2[e]
// Done as a single V_WMMA per wave:
//   A[m,k] = W2[e_base+m, k]      (16x4 f32 tile, matches ISA A layout)
//   B[k,n] = qscaled              (constant -> layout-irrelevant)
//   C[m,n] = b2[e_base+m]         (broadcast along N, matches C layout)
//   D[m,n] = row[e_base+m]        (duplicated along N; lane 0 / lane 16 write)
// 128 waves cover all 2048 columns of E.
// ---------------------------------------------------------------------------
__global__ void qffn_row_wmma(const float* __restrict__ W2,
                              const float* __restrict__ b2,
                              const float* __restrict__ qp,
                              float* __restrict__ row) {
  const int lane  = threadIdx.x & 31;
  const int wave  = blockIdx.x * (blockDim.x >> 5) + (threadIdx.x >> 5);
  const int e_base = wave * 16;

  const int m     = lane & 15;   // row within tile (A) / N column (C/D)
  const int khalf = lane >> 4;   // 0: K=0,1 / M=0..7 half; 1: K=2,3 / M=8..15 half

  // relu(cos(q_params[0,0] + q_params[0,1]))
  const float qs = fmaxf(cosf(qp[0] + qp[1]), 0.0f);

  // C: lane needs b2[e_base + khalf*8 + 0..7] (M varies over VGPR index)
  const float* crow = b2 + e_base + khalf * 8;
  v8f c;
#pragma unroll
  for (int v = 0; v < 8; ++v) c[v] = crow[v];

  v8f d;
#if __has_builtin(__builtin_amdgcn_wmma_f32_16x16x4_f32)
  // A layout (32-bit 16x4): VGPR0 = K0 (lanes 0-15) / K2 (lanes 16-31),
  //                          VGPR1 = K1 / K3.
  const float* arow = W2 + (size_t)(e_base + m) * FF_DIM + khalf * 2;
  v2f a;
  a.x = arow[0];
  a.y = arow[1];
  v2f b;
  b.x = qs;
  b.y = qs;
  d = __builtin_amdgcn_wmma_f32_16x16x4_f32(
      /*neg_a=*/false, a, /*neg_b=*/false, b,
      /*c_mod=*/(short)0, c, /*reuse_a=*/false, /*reuse_b=*/false);
#else
  // Fallback: f16 WMMA (codegen-confirmed). Zero-pad K beyond 4; f16 holds
  // W2 (~0.02 magnitude) with ample precision for this reduction.
  v16h a16 = (v16h)(_Float16)0.0f;
  if (khalf == 0) {
    // lanes 0-15 hold K=0..7 in VGPRs 0..3 (two halves per VGPR)
    const float* arow = W2 + (size_t)(e_base + m) * FF_DIM;
    a16[0] = (_Float16)arow[0];
    a16[1] = (_Float16)arow[1];
    a16[2] = (_Float16)arow[2];
    a16[3] = (_Float16)arow[3];
  }
  v16h b16;
#pragma unroll
  for (int v = 0; v < 16; ++v) b16[v] = (_Float16)qs;
  d = __builtin_amdgcn_wmma_f32_16x16x32_f16(
      false, a16, false, b16, (short)0, c, false, false);
#endif

  // D: lane holds d[v] = row[e_base + v + khalf*8], duplicated across 16 lanes
  // per half. One writer per half-wave.
  if ((lane & 15) == 0) {
    float* r = row + e_base + khalf * 8;
#pragma unroll
    for (int v = 0; v < 8; ++v) r[v] = d[v];
  }
}

// ---------------------------------------------------------------------------
// Phase 2: broadcast the 2048-float row to all 8192 output rows.
// Store-bandwidth bound: 64 MiB -> ~2.9 us at 23.3 TB/s.
// Each thread caches one 16B chunk of the row in VGPRs, then streams 8
// fully-coalesced b128 stores (consecutive lanes -> contiguous memory).
// ---------------------------------------------------------------------------
#define ROW_CHUNKS (E_DIM / 4)         // 512 float4 chunks per row
#define ROWS_PER_THREAD 8

__global__ void qffn_broadcast(const float* __restrict__ row,
                               float* __restrict__ out) {
  const int g    = blockIdx.x * blockDim.x + threadIdx.x;
  const int col4 = g & (ROW_CHUNKS - 1);   // chunk within row
  const int rg   = g >> 9;                 // row group (1024 groups)

  const v4f val = ((const v4f*)row)[col4];

  float* base = out + (size_t)rg * ROWS_PER_THREAD * E_DIM + col4 * 4;
#pragma unroll
  for (int r = 0; r < ROWS_PER_THREAD; ++r) {
    *(v4f*)(base + (size_t)r * E_DIM) = val;
  }
}

// ---------------------------------------------------------------------------
// Inputs (setup_inputs order): x, W1, b1, W2, b2, q_params  (all f32)
// Output: (B, E) f32
// ---------------------------------------------------------------------------
extern "C" void kernel_launch(void* const* d_in, const int* in_sizes, int n_in,
                              void* d_out, int out_size, void* d_ws, size_t ws_size,
                              hipStream_t stream) {
  const float* W2 = (const float*)d_in[3];
  const float* b2 = (const float*)d_in[4];
  const float* qp = (const float*)d_in[5];
  float* out = (float*)d_out;
  float* row = (float*)d_ws;   // 2048 floats of scratch

  // 16 blocks * 8 waves = 128 waves -> 128 * 16 = 2048 columns
  qffn_row_wmma<<<16, 256, 0, stream>>>(W2, b2, qp, row);

  // 512 chunks * 1024 row-groups = 524288 threads -> 2048 blocks of 256
  const int total = ROW_CHUNKS * (B_DIM / ROWS_PER_THREAD);
  qffn_broadcast<<<total / 256, 256, 0, stream>>>(row, out);
}
